// LeViTBlock_48833778155732
// MI455X (gfx1250) — compile-verified, matrix operand order checked by reading, not run on hardware
//
#include <hip/hip_runtime.h>
#include <hip/hip_bf16.h>
#include <cstddef>

typedef __bf16 bfx;
typedef __attribute__((ext_vector_type(8)))  __bf16 v8bf;
typedef __attribute__((ext_vector_type(16))) __bf16 v16bf;
typedef __attribute__((ext_vector_type(8)))  float  v8f;

__device__ __forceinline__ float hswish(float v) {
    return v * fminf(fmaxf(v + 3.0f, 0.0f), 6.0f) * (1.0f / 6.0f);
}

// ---------------------------------------------------------------------------
// Elementwise helpers
// ---------------------------------------------------------------------------
__global__ void k_cast_bf16(const float* __restrict__ s, bfx* __restrict__ d, int n) {
    int i = blockIdx.x * blockDim.x + threadIdx.x;
    if (i < n) d[i] = (bfx)s[i];
}

__global__ void k_dual(const float* __restrict__ s, float* __restrict__ df,
                       bfx* __restrict__ db, int n) {
    int i = blockIdx.x * blockDim.x + threadIdx.x;
    if (i < n) { float v = s[i]; df[i] = v; db[i] = (bfx)v; }
}

// subsample gather: x[B,14,14,C][:, ::2, ::2] -> xq[B,49,C]   (bf16)
__global__ void k_gather(const bfx* __restrict__ xb, bfx* __restrict__ xq, int n) {
    int i = blockIdx.x * blockDim.x + threadIdx.x;
    if (i >= n) return;
    int c = i & 255;
    int r = i >> 8;
    int b = r / 49;
    int q = r - b * 49;
    int qi = q / 7, qj = q - qi * 7;
    int sr = b * 196 + qi * 28 + qj * 2;
    xq[i] = xb[(size_t)sr * 256 + c];
}

// ---------------------------------------------------------------------------
// Fused GEMM:  out[m,n] = epi( sum_k A[m,k]*W[n,k] * sc[n] + tb[n] )
//   epi: optional hardswish, optional += res (fp32), write fp32 and/or bf16.
// A: [M,K] bf16 row-major, W: [N,K] bf16 row-major (i.e. X @ W^T).
// Block: 256 thr = 8 waves; block tile 128(M) x 64(N); wave tile 32 x 32
// (2 A-frags x 2 B-frags -> 4 WMMAs per K-step, 16 FLOP/byte from cache).
// M % 128 == 0, N % 64 == 0, K % 32 == 0 for all call sites here.
// ---------------------------------------------------------------------------
__global__ __launch_bounds__(256) void k_gemm(
    const bfx* __restrict__ A, const bfx* __restrict__ W,
    const float* __restrict__ sc, const float* __restrict__ tb,
    const float* __restrict__ res, float* __restrict__ outf,
    bfx* __restrict__ outb, int M, int N, int K, int act)
{
    int tid  = threadIdx.x;
    int wave = tid >> 5;
    int lane = tid & 31;
    int hlf  = lane >> 4;      // which 16-lane half
    int l16  = lane & 15;
    int wm   = wave & 3;       // 4 M sub-tiles of 32
    int wn   = wave >> 2;      // 2 N sub-tiles of 32
    int mbase = blockIdx.y * 128 + wm * 32;
    int nbase = blockIdx.x * 64 + wn * 32;

    // A fragment: lane (l16,hlf) holds row (mbase+l16); elems 0..7 = K+8*hlf..+7,
    // elems 8..15 = K+16+8*hlf..+7  (per CDNA5 16-bit A 16x32 layout).
    const bfx* ap0 = A + (size_t)(mbase + l16) * K + 8 * hlf;
    const bfx* ap1 = A + (size_t)(mbase + 16 + l16) * K + 8 * hlf;
    const bfx* b0p = W + (size_t)(nbase + l16) * K + 8 * hlf;
    const bfx* b1p = W + (size_t)(nbase + 16 + l16) * K + 8 * hlf;

    v8f c00 = {};
    v8f c01 = {};
    v8f c10 = {};
    v8f c11 = {};

    for (int k = 0; k < K; k += 32) {
        v8bf al0 = *(const v8bf*)(ap0 + k);
        v8bf ah0 = *(const v8bf*)(ap0 + k + 16);
        v16bf a0 = __builtin_shufflevector(al0, ah0, 0,1,2,3,4,5,6,7,8,9,10,11,12,13,14,15);

        v8bf al1 = *(const v8bf*)(ap1 + k);
        v8bf ah1 = *(const v8bf*)(ap1 + k + 16);
        v16bf a1 = __builtin_shufflevector(al1, ah1, 0,1,2,3,4,5,6,7,8,9,10,11,12,13,14,15);

        v8bf bl0 = *(const v8bf*)(b0p + k);
        v8bf bh0 = *(const v8bf*)(b0p + k + 16);
        v16bf b0 = __builtin_shufflevector(bl0, bh0, 0,1,2,3,4,5,6,7,8,9,10,11,12,13,14,15);

        v8bf bl1 = *(const v8bf*)(b1p + k);
        v8bf bh1 = *(const v8bf*)(b1p + k + 16);
        v16bf b1 = __builtin_shufflevector(bl1, bh1, 0,1,2,3,4,5,6,7,8,9,10,11,12,13,14,15);

        __builtin_prefetch(ap0 + k + 32, 0, 1);
        __builtin_prefetch(b0p + k + 32, 0, 1);

        c00 = __builtin_amdgcn_wmma_f32_16x16x32_bf16(false, a0, false, b0,
                                                      (short)0, c00, false, false);
        c01 = __builtin_amdgcn_wmma_f32_16x16x32_bf16(false, a0, false, b1,
                                                      (short)0, c01, false, false);
        c10 = __builtin_amdgcn_wmma_f32_16x16x32_bf16(false, a1, false, b0,
                                                      (short)0, c10, false, false);
        c11 = __builtin_amdgcn_wmma_f32_16x16x32_bf16(false, a1, false, b1,
                                                      (short)0, c11, false, false);
    }

    // C/D layout: VGPR r -> row (+r + 8*hlf), col (+l16)
    int n0 = nbase + l16;
    int n1 = nbase + 16 + l16;
    float s0 = sc[n0], t0 = tb[n0];
    float s1 = sc[n1], t1 = tb[n1];
    int mr0 = mbase + 8 * hlf;        // rows of A-frag 0 accumulators
    int mr1 = mbase + 16 + 8 * hlf;   // rows of A-frag 1 accumulators
#pragma unroll
    for (int r = 0; r < 8; ++r) {
        size_t m0 = (size_t)(mr0 + r);
        size_t m1 = (size_t)(mr1 + r);
        float y00 = c00[r] * s0 + t0;
        float y01 = c01[r] * s1 + t1;
        float y10 = c10[r] * s0 + t0;
        float y11 = c11[r] * s1 + t1;
        if (act) {
            y00 = hswish(y00); y01 = hswish(y01);
            y10 = hswish(y10); y11 = hswish(y11);
        }
        if (res) {
            y00 += res[m0 * N + n0]; y01 += res[m0 * N + n1];
            y10 += res[m1 * N + n0]; y11 += res[m1 * N + n1];
        }
        if (outf) {
            outf[m0 * N + n0] = y00; outf[m0 * N + n1] = y01;
            outf[m1 * N + n0] = y10; outf[m1 * N + n1] = y11;
        }
        if (outb) {
            outb[m0 * N + n0] = (bfx)y00; outb[m0 * N + n1] = (bfx)y01;
            outb[m1 * N + n0] = (bfx)y10; outb[m1 * N + n1] = (bfx)y11;
        }
    }
}

// ---------------------------------------------------------------------------
// Attention: per (b,h) block. q: [Sq, 32] bf16 rows at qbuf + h*q_hs,
// k/v packed per token at kvbuf + h*kv_hs (+k_add / +v_add). Relative bias
// index computed analytically: idx = |q_mul*qi - ki|*14 + |q_mul*qj - kj|.
// Output: hardswish(softmax(qk^T*scale + bias) @ v)  -> bf16 at h*o_hs.
// blockDim = 256 (8 waves); one q row per wave per iteration.
// ---------------------------------------------------------------------------
__global__ __launch_bounds__(256) void k_attn(
    const bfx* __restrict__ qbuf, int q_rs, int q_hs,
    const bfx* __restrict__ kvbuf, int kv_rs, int kv_hs, int k_add, int v_add,
    const float* __restrict__ bias,
    bfx* __restrict__ obuf, int o_rs, int o_hs,
    int H, int Sq, int Skv, int VD, float scale, int q_w, int q_mul)
{
    __shared__ float sp[8][208];

    int b = blockIdx.x / H;
    int h = blockIdx.x - b * H;
    int w    = threadIdx.x >> 5;
    int lane = threadIdx.x & 31;

    const bfx* qbase = qbuf + (size_t)b * Sq * q_rs + h * q_hs;
    const bfx* kbase = kvbuf + (size_t)b * Skv * kv_rs + h * kv_hs + k_add;
    const bfx* vbase = kvbuf + (size_t)b * Skv * kv_rs + h * kv_hs + v_add;
    const float* bh  = bias + h * 196;

    int iters = (Sq + 7) >> 3;
    for (int it = 0; it < iters; ++it) {
        int qrow = it * 8 + w;
        bool active = qrow < Sq;
        float inv = 0.0f;

        if (active) {
            const bfx* qp = qbase + (size_t)qrow * q_rs;
            v8bf q0 = *(const v8bf*)(qp);
            v8bf q1 = *(const v8bf*)(qp + 8);
            v8bf q2 = *(const v8bf*)(qp + 16);
            v8bf q3 = *(const v8bf*)(qp + 24);
            float qf[32];
#pragma unroll
            for (int i = 0; i < 8; ++i) {
                qf[i]      = (float)q0[i];
                qf[8 + i]  = (float)q1[i];
                qf[16 + i] = (float)q2[i];
                qf[24 + i] = (float)q3[i];
            }
            int qi = qrow / q_w, qj = qrow - qi * q_w;

            float lg[7];
            float mx = -1e30f;
#pragma unroll
            for (int i = 0; i < 7; ++i) {
                int k = lane + i * 32;
                if (k < Skv) {
                    const bfx* kp = kbase + (size_t)k * kv_rs;
                    v8bf k0 = *(const v8bf*)(kp);
                    v8bf k1 = *(const v8bf*)(kp + 8);
                    v8bf k2 = *(const v8bf*)(kp + 16);
                    v8bf k3 = *(const v8bf*)(kp + 24);
                    float acc = 0.0f;
#pragma unroll
                    for (int j = 0; j < 8; ++j) acc += qf[j]      * (float)k0[j];
#pragma unroll
                    for (int j = 0; j < 8; ++j) acc += qf[8 + j]  * (float)k1[j];
#pragma unroll
                    for (int j = 0; j < 8; ++j) acc += qf[16 + j] * (float)k2[j];
#pragma unroll
                    for (int j = 0; j < 8; ++j) acc += qf[24 + j] * (float)k3[j];
                    int ki = k / 14, kj = k - ki * 14;
                    int di = q_mul * qi - ki; if (di < 0) di = -di;
                    int dj = q_mul * qj - kj; if (dj < 0) dj = -dj;
                    float lv = acc * scale + bh[di * 14 + dj];
                    lg[i] = lv;
                    mx = fmaxf(mx, lv);
                }
            }
#pragma unroll
            for (int o = 16; o > 0; o >>= 1) mx = fmaxf(mx, __shfl_xor(mx, o, 32));

            float sum = 0.0f;
#pragma unroll
            for (int i = 0; i < 7; ++i) {
                int k = lane + i * 32;
                if (k < Skv) {
                    float e = __expf(lg[i] - mx);
                    sp[w][k] = e;
                    sum += e;
                }
            }
#pragma unroll
            for (int o = 16; o > 0; o >>= 1) sum += __shfl_xor(sum, o, 32);
            inv = 1.0f / sum;
        }
        __syncthreads();

        if (active) {
            for (int dd = lane; dd < VD; dd += 32) {
                const bfx* vp = vbase + dd;
                float acc = 0.0f;
                for (int k = 0; k < Skv; ++k)
                    acc += sp[w][k] * (float)vp[(size_t)k * kv_rs];
                float o = hswish(acc * inv);
                obuf[((size_t)b * Sq + qrow) * o_rs + h * o_hs + dd] = (bfx)o;
            }
        }
        __syncthreads();
    }
}

// ---------------------------------------------------------------------------
// Host launcher
// ---------------------------------------------------------------------------
extern "C" void kernel_launch(void* const* d_in, const int* in_sizes, int n_in,
                              void* d_out, int out_size, void* d_ws, size_t ws_size,
                              hipStream_t stream)
{
    (void)in_sizes; (void)n_in; (void)out_size; (void)ws_size;

    const float* x      = (const float*)d_in[0];
    const float* Wqkv   = (const float*)d_in[1];
    const float* qkv_s  = (const float*)d_in[2];
    const float* qkv_t  = (const float*)d_in[3];
    const float* Wproj  = (const float*)d_in[4];
    const float* proj_s = (const float*)d_in[5];
    const float* proj_t = (const float*)d_in[6];
    const float* abias  = (const float*)d_in[7];
    const float* W1     = (const float*)d_in[8];
    const float* m1s    = (const float*)d_in[9];
    const float* m1t    = (const float*)d_in[10];
    const float* W2     = (const float*)d_in[11];
    const float* m2s    = (const float*)d_in[12];
    const float* m2t    = (const float*)d_in[13];
    const float* sWkv   = (const float*)d_in[14];
    const float* skvs   = (const float*)d_in[15];
    const float* skvt   = (const float*)d_in[16];
    const float* sWq    = (const float*)d_in[17];
    const float* sqs    = (const float*)d_in[18];
    const float* sqt    = (const float*)d_in[19];
    const float* sWp    = (const float*)d_in[20];
    const float* sps    = (const float*)d_in[21];
    const float* spt    = (const float*)d_in[22];
    const float* sbias  = (const float*)d_in[23];
    const float* sm1    = (const float*)d_in[24];
    const float* sm1s   = (const float*)d_in[25];
    const float* sm1t   = (const float*)d_in[26];
    const float* sm2    = (const float*)d_in[27];
    const float* sm2s   = (const float*)d_in[28];
    const float* sm2t   = (const float*)d_in[29];

    const int B  = 256, S = 196, C = 256;
    const int M  = B * S;   // 50176 (div by 128)
    const int M2 = B * 49;  // 12544 (div by 128)

    char* ws = (char*)d_ws;
    size_t off = 0;
    auto carve = [&](size_t bytes) -> void* {
        void* p = ws + off;
        off = (off + bytes + 255) & ~(size_t)255;
        return p;
    };

    float* xf     = (float*)carve((size_t)M  * C    * 4);
    bfx*   xb     = (bfx*)  carve((size_t)M  * C    * 2);
    bfx*   buf512 = (bfx*)  carve((size_t)M  * 512  * 2);  // qkv, then mlp hidden
    bfx*   ob     = (bfx*)  carve((size_t)M  * 256  * 2);  // attn out (also sub o: 12544*1024)
    bfx*   kvb    = (bfx*)  carve((size_t)M  * 1280 * 2);
    bfx*   xqb    = (bfx*)  carve((size_t)M2 * 256  * 2);
    bfx*   qb     = (bfx*)  carve((size_t)M2 * 256  * 2);
    float* x2f    = (float*)carve((size_t)M2 * 384  * 4);
    bfx*   x2b    = (bfx*)  carve((size_t)M2 * 384  * 2);
    bfx*   h2b    = (bfx*)  carve((size_t)M2 * 768  * 2);
    bfx*   wb     = (bfx*)  carve((size_t)3211264   * 2);

    bfx* wb_qkv  = wb;
    bfx* wb_proj = wb_qkv  + 524288;
    bfx* wb_m1   = wb_proj + 262144;
    bfx* wb_m2   = wb_m1   + 524288;
    bfx* wb_skv  = wb_m2   + 524288;
    bfx* wb_sq   = wb_skv  + 327680;
    bfx* wb_sp   = wb_sq   + 65536;
    bfx* wb_sm1  = wb_sp   + 393216;
    bfx* wb_sm2  = wb_sm1  + 294912;

    auto cast = [&](const float* src, bfx* dst, int n) {
        k_cast_bf16<<<dim3((n + 255) / 256), dim3(256), 0, stream>>>(src, dst, n);
    };
    cast(Wqkv,  wb_qkv,  524288);
    cast(Wproj, wb_proj, 262144);
    cast(W1,    wb_m1,   524288);
    cast(W2,    wb_m2,   524288);
    cast(sWkv,  wb_skv,  327680);
    cast(sWq,   wb_sq,   65536);
    cast(sWp,   wb_sp,   393216);
    cast(sm1,   wb_sm1,  294912);
    cast(sm2,   wb_sm2,  294912);

    k_dual<<<dim3((M * C + 255) / 256), dim3(256), 0, stream>>>(x, xf, xb, M * C);

    auto gemm = [&](const bfx* A, const bfx* Wm, const float* scv, const float* tbv,
                    const float* rsv, float* of, bfx* obp, int Mm, int Nn, int Kk, int act) {
        k_gemm<<<dim3(Nn / 64, Mm / 128), dim3(256), 0, stream>>>(
            A, Wm, scv, tbv, rsv, of, obp, Mm, Nn, Kk, act);
    };

    const float scale = 0.17677669529663687f;  // 1/sqrt(32)

    for (int l = 0; l < 4; ++l) {
        // qkv = x @ Wqkv^T * s + t  -> bf16 [M,512]
        gemm(xb, wb_qkv + (size_t)l * 512 * 256, qkv_s + l * 512, qkv_t + l * 512,
             nullptr, nullptr, buf512, M, 512, 256, 0);
        // attention (H1=4, KD=32, VD=64) -> hardswish(o) bf16 [M,256]
        k_attn<<<dim3(B * 4), dim3(256), 0, stream>>>(
            buf512, 512, 128,
            buf512, 512, 128, 32, 64,
            abias + (size_t)l * 4 * 196,
            ob, 256, 64,
            4, S, S, 64, scale, 14, 1);
        // x = x + hardswish(o) @ Wproj^T * s + t   (fp32 + bf16 out)
        gemm(ob, wb_proj + (size_t)l * 256 * 256, proj_s + l * 256, proj_t + l * 256,
             xf, xf, xb, M, 256, 256, 0);
        // h = hardswish(x @ W1^T * s + t)  -> bf16 [M,512]
        gemm(xb, wb_m1 + (size_t)l * 512 * 256, m1s + l * 512, m1t + l * 512,
             nullptr, nullptr, buf512, M, 512, 256, 1);
        // x = x + h @ W2^T * s + t
        gemm(buf512, wb_m2 + (size_t)l * 256 * 512, m2s + l * 256, m2t + l * 256,
             xf, xf, xb, M, 256, 512, 0);
    }

    // ----- subsample stage -----
    gemm(xb, wb_skv, skvs, skvt, nullptr, nullptr, kvb, M, 1280, 256, 0);
    k_gather<<<dim3((M2 * 256 + 255) / 256), dim3(256), 0, stream>>>(xb, xqb, M2 * 256);
    gemm(xqb, wb_sq, sqs, sqt, nullptr, nullptr, qb, M2, 256, 256, 0);
    // sub attention (H2=8, KD=32, VD=128), strided bias -> hardswish(o) bf16 [M2,1024]
    k_attn<<<dim3(B * 8), dim3(256), 0, stream>>>(
        qb, 256, 32,
        kvb, 1280, 160, 0, 32,
        sbias,
        ob, 1024, 128,
        8, 49, S, 128, scale, 7, 2);
    // x2 = hardswish(o) @ sub_Wproj^T * s + t
    gemm(ob, wb_sp, sps, spt, nullptr, x2f, x2b, M2, 384, 1024, 0);
    // h2 = hardswish(x2 @ smlp_W1^T * s + t)
    gemm(x2b, wb_sm1, sm1s, sm1t, nullptr, nullptr, h2b, M2, 768, 384, 1);
    // out = x2 + h2 @ smlp_W2^T * s + t   (fp32 -> d_out)
    gemm(h2b, wb_sm2, sm2s, sm2t, x2f, (float*)d_out, nullptr, M2, 384, 768, 0);
}
